// MambaLayer_20083267076883
// MI455X (gfx1250) — compile-verified, matrix operand order checked by reading, not run on hardware
//
#include <hip/hip_runtime.h>

// ---------------- problem constants ----------------
#define BB   2
#define CC   128
#define LL   4096            // 16*16*16
#define MM   (BB*LL)         // 8192 tokens
#define DIN  256             // d_inner
#define NS   16              // state dim N
#define RK   8               // dt_rank
#define XPAD 64              // x_proj out rows padded 40 -> 64

// ---------------- CDNA5 feature detection ----------------
#if defined(__AMDGCN__) && __has_builtin(__builtin_amdgcn_global_load_async_to_lds_b128) && \
    __has_builtin(__builtin_amdgcn_s_wait_asynccnt)
#define ASYNC_LDS 1
#else
#define ASYNC_LDS 0
#endif

// ---------------- WMMA types ----------------
typedef __attribute__((ext_vector_type(16))) __bf16       v16bf;
typedef __attribute__((ext_vector_type(8)))  float        v8f;
typedef __attribute__((ext_vector_type(4)))  unsigned int v4u;

union FU { v4u q[2]; v16bf bf; };

#if ASYNC_LDS
typedef int v4i __attribute__((ext_vector_type(4)));
typedef __attribute__((address_space(1))) v4i v4i_gas;   // global int4*
typedef __attribute__((address_space(3))) v4i v4i_las;   // LDS int4*

// 16-byte async copy global -> LDS (GLOBAL_LOAD_ASYNC_TO_LDS_B128, tracked by ASYNCcnt).
// AS pointers built via exact-width integer casts: AS1 is 64-bit, AS3 is 32-bit
// (low 32 bits of a generic LDS-aperture address are the LDS byte offset).
__device__ __forceinline__ void async_cp16(const float* g, float* l) {
    __builtin_amdgcn_global_load_async_to_lds_b128(
        (v4i_gas*)(unsigned long long)(__UINTPTR_TYPE__)g,
        (v4i_las*)(unsigned int)(__UINTPTR_TYPE__)l, 0, 0);
}
#endif

__device__ __forceinline__ unsigned short f2bf(float f) {
    unsigned int u = __builtin_bit_cast(unsigned int, f);
    u += 0x7FFFu + ((u >> 16) & 1u);               // round-to-nearest-even
    return (unsigned short)(u >> 16);
}
__device__ __forceinline__ float bf2f(unsigned short h) {
    unsigned int u = ((unsigned int)h) << 16;
    return __builtin_bit_cast(float, u);
}
__device__ __forceinline__ float siluf(float z)     { return z / (1.f + __expf(-z)); }
__device__ __forceinline__ float softplusf(float x) { return (x > 20.f) ? x : log1pf(__expf(x)); }

// fragment load per ISA 16-bit A/B layout: two 16B chunks at ushort offsets base and 16+base
__device__ __forceinline__ void ldfrag(const unsigned short* p, int base, FU& f) {
    f.q[0] = *(const v4u*)(p + base);
    f.q[1] = *(const v4u*)(p + 16 + base);
}

#define WMMA_BF16(Af, Bf, Cacc) \
    __builtin_amdgcn_wmma_f32_16x16x32_bf16(false, (Af).bf, false, (Bf).bf, (short)0, (Cacc), false, false)

// ---------------- kernels ----------------

// x [B,C,L] f32  ->  xt [(b*L+l)*C + c] bf16
__global__ void k_x_transpose(const float* __restrict__ x, unsigned short* __restrict__ xt) {
    int idx = blockIdx.x * blockDim.x + threadIdx.x;
    if (idx >= BB * CC * LL) return;
    int b = idx / (CC * LL);
    int r = idx - b * CC * LL;
    int c = r / LL;
    int l = r - c * LL;
    xt[((size_t)(b * LL + l)) * CC + c] = f2bf(x[idx]);
}

// generic weight convert with optional flips / zero-padding
__global__ void k_wconv(const float* __restrict__ src, unsigned short* __restrict__ dst,
                        int rows, int cols, int dstRows, int flipR, int flipC) {
    int idx = blockIdx.x * blockDim.x + threadIdx.x;
    if (idx >= dstRows * cols) return;
    int r = idx / cols, c = idx - r * cols;
    float v = 0.f;
    if (r < rows) {
        int sr = flipR ? (rows - 1 - r) : r;
        int sc = flipC ? (cols - 1 - c) : c;
        v = src[(size_t)sr * cols + sc];
    }
    dst[idx] = f2bf(v);
}

// Out[m,n] (+)= sum_k A[m,k]*W[n,k].  Wave computes a 32x32 tile (2x2 WMMA register block),
// block = 8 waves as 4(M) x 2(N) -> 128x64 block tile.  Double-buffered K loop (step 32).
__global__ void k_gemm_bf16(const unsigned short* __restrict__ A,
                            const unsigned short* __restrict__ W,
                            float* __restrict__ O,
                            int M, int Npad, int K, int ldo, int beta) {
    int wave = threadIdx.x >> 5;
    int lane = threadIdx.x & 31;
    int tm = blockIdx.x * 128 + (wave >> 1) * 32;
    int tn = blockIdx.y * 64  + (wave &  1) * 32;
    if (tm >= M || tn >= Npad) return;                 // wave-uniform: EXEC all-1 for WMMA
    int row  = lane & 15;
    int base = (lane >> 4) << 3;
    const unsigned short* a0 = A + (size_t)(tm + row) * K;
    const unsigned short* a1 = a0 + (size_t)16 * K;
    const unsigned short* b0 = W + (size_t)(tn + row) * K;
    const unsigned short* b1 = b0 + (size_t)16 * K;

    v8f c00 = {}, c01 = {}, c10 = {}, c11 = {};
    FU A0[2], A1[2], B0[2], B1[2];
    ldfrag(a0, base, A0[0]); ldfrag(a1, base, A1[0]);
    ldfrag(b0, base, B0[0]); ldfrag(b1, base, B1[0]);

    int kc = 0;
    while (true) {
        int kn = kc + 32;
        if (kn < K) {                                   // prefetch chunk kn into buffer 1
            ldfrag(a0 + kn, base, A0[1]); ldfrag(a1 + kn, base, A1[1]);
            ldfrag(b0 + kn, base, B0[1]); ldfrag(b1 + kn, base, B1[1]);
        }
        c00 = WMMA_BF16(A0[0], B0[0], c00);
        c01 = WMMA_BF16(A0[0], B1[0], c01);
        c10 = WMMA_BF16(A1[0], B0[0], c10);
        c11 = WMMA_BF16(A1[0], B1[0], c11);
        kc = kn;
        if (kc >= K) break;

        kn = kc + 32;
        if (kn < K) {                                   // prefetch chunk kn into buffer 0
            ldfrag(a0 + kn, base, A0[0]); ldfrag(a1 + kn, base, A1[0]);
            ldfrag(b0 + kn, base, B0[0]); ldfrag(b1 + kn, base, B1[0]);
        }
        c00 = WMMA_BF16(A0[1], B0[1], c00);
        c01 = WMMA_BF16(A0[1], B1[1], c01);
        c10 = WMMA_BF16(A1[1], B0[1], c10);
        c11 = WMMA_BF16(A1[1], B1[1], c11);
        kc = kn;
        if (kc >= K) break;
    }

    int n0 = tn + row, n1 = n0 + 16;
#pragma unroll
    for (int r = 0; r < 8; ++r) {
        int m0 = tm + r + base, m1 = m0 + 16;           // C/D layout: lanes16-31 -> M+8
        size_t i00 = (size_t)m0 * ldo + n0, i01 = (size_t)m0 * ldo + n1;
        size_t i10 = (size_t)m1 * ldo + n0, i11 = (size_t)m1 * ldo + n1;
        float v00 = c00[r], v01 = c01[r], v10 = c10[r], v11 = c11[r];
        if (beta) { v00 += O[i00]; v01 += O[i01]; v10 += O[i10]; v11 += O[i11]; }
        O[i00] = v00; O[i01] = v01; O[i10] = v10; O[i11] = v11;
    }
}

// BN1(eval)+ReLU epilogue: h_t f32 [M,C] -> act_t f32 and x_flat bf16
__global__ void k_bn1_relu(const float* __restrict__ ht,
                           const float* __restrict__ g, const float* __restrict__ bta,
                           const float* __restrict__ mu, const float* __restrict__ var,
                           float* __restrict__ act, unsigned short* __restrict__ xflat) {
    int idx = blockIdx.x * blockDim.x + threadIdx.x;
    if (idx >= MM * CC) return;
    int c = idx & (CC - 1);
    float s = g[c] * rsqrtf(var[c] + 1e-5f);
    float v = (ht[idx] - mu[c]) * s + bta[c];
    v = fmaxf(v, 0.f);
    act[idx] = v;
    xflat[idx] = f2bf(v);
}

// depthwise causal conv1d (direction-aware) + SiLU
__global__ void k_conv_silu(const float* __restrict__ xz, const float* __restrict__ cw,
                            const float* __restrict__ cb, unsigned short* __restrict__ xm,
                            int fL) {
    int idx = blockIdx.x * blockDim.x + threadIdx.x;
    if (idx >= MM * DIN) return;
    int d = idx & (DIN - 1);
    int l = (idx / DIN) & (LL - 1);
    int b = idx / (DIN * LL);
    size_t rowb = (size_t)b * LL;
    float acc = cb[d];
#pragma unroll
    for (int j = 0; j < 4; ++j) {
        int ll = fL ? (l + 3 - j) : (l - 3 + j);
        if (ll >= 0 && ll < LL)
            acc = fmaf(cw[d * 4 + j], xz[(rowb + ll) * 512 + d], acc);
    }
    xm[idx] = f2bf(siluf(acc));
}

// selective scan, fused dt_proj + softplus + recurrence + D skip + SiLU(z) gate.
// one block per batch; thread = channel; per-step dt/B/C row staged in LDS.
// CDNA5 path: next row fetched with GLOBAL_LOAD_ASYNC_TO_LDS_B128 (ASYNCcnt double buffer).
__global__ void k_scan(const float* __restrict__ xdbl,      // [M, XPAD]
                       const unsigned short* __restrict__ xm,
                       const float* __restrict__ xz,        // z at cols [DIN, 2*DIN)
                       const float* __restrict__ dtw, const float* __restrict__ dtb,
                       const float* __restrict__ Alog, const float* __restrict__ Dp,
                       unsigned short* __restrict__ yg, int fL) {
    __shared__ float sh[2][64];
    int b = blockIdx.x;
    int d = threadIdx.x;
    float Arow[NS], h[NS], wdt[RK];
#pragma unroll
    for (int n = 0; n < NS; ++n) { Arow[n] = -__expf(Alog[d * NS + n]); h[n] = 0.f; }
#pragma unroll
    for (int r = 0; r < RK; ++r) wdt[r] = dtw[d * RK + r];
    float bias = dtb[d], Dd = Dp[d];
    size_t rowb = (size_t)b * LL;
    int wave0 = (threadIdx.x >> 5) == 0;

#if ASYNC_LDS
    // prologue: async-copy row for step 0 into buffer 0 (12 lanes x 16B = 48 floats)
    {
        int l0 = fL ? (LL - 1) : 0;
        if (threadIdx.x < 12)
            async_cp16(xdbl + (rowb + l0) * XPAD + 4 * threadIdx.x,
                       &sh[0][4 * threadIdx.x]);
        if (wave0) __builtin_amdgcn_s_wait_asynccnt(0);
    }
    __syncthreads();
    int cur = 0;
    for (int s = 0; s < LL; ++s) {
        int nxt = cur ^ 1;
        if (s + 1 < LL) {
            int ln = fL ? (LL - 2 - s) : (s + 1);
            if (threadIdx.x < 12)
                async_cp16(xdbl + (rowb + ln) * XPAD + 4 * threadIdx.x,
                           &sh[nxt][4 * threadIdx.x]);
            if (wave0) __builtin_amdgcn_s_wait_asynccnt(1);   // row s complete, next in flight
        } else if (wave0) {
            __builtin_amdgcn_s_wait_asynccnt(0);
        }
        __syncthreads();
        const float* row_sh = sh[cur];
#else
    for (int s = 0; s < LL; ++s) {
        int l0 = fL ? (LL - 1 - s) : s;
        if (threadIdx.x < 48) sh[0][threadIdx.x] = xdbl[(rowb + l0) * XPAD + threadIdx.x];
        __syncthreads();
        const float* row_sh = sh[0];
#endif
        int l = fL ? (LL - 1 - s) : s;
        size_t row = rowb + l;
        float draw = bias;
#pragma unroll
        for (int r = 0; r < RK; ++r) draw = fmaf(row_sh[r], wdt[r], draw);
        float delta = softplusf(draw);
        float u  = bf2f(xm[row * DIN + d]);
        float du = delta * u;
        float y  = 0.f;
#pragma unroll
        for (int n = 0; n < NS; ++n) {
            float dA = __expf(delta * Arow[n]);
            h[n] = fmaf(dA, h[n], du * row_sh[8 + n]);
            y    = fmaf(h[n], row_sh[24 + n], y);
        }
        y = fmaf(u, Dd, y);
        float z = xz[row * 512 + DIN + d];
        yg[row * DIN + d] = f2bf(y * siluf(z));
        __syncthreads();
#if ASYNC_LDS
        cur = nxt;
#endif
    }
    (void)wave0;
}

// mid = ysum/4 + act  (bf16)
__global__ void k_mid(const float* __restrict__ ysum, const float* __restrict__ act,
                      unsigned short* __restrict__ mid) {
    int idx = blockIdx.x * blockDim.x + threadIdx.x;
    if (idx >= MM * CC) return;
    mid[idx] = f2bf(fmaf(ysum[idx], 0.25f, act[idx]));
}

// BN2(eval)+ReLU with transposed store: tmp [M,C] -> out [B,C,L]
__global__ void k_bn2_out(const float* __restrict__ t,
                          const float* __restrict__ g, const float* __restrict__ bta,
                          const float* __restrict__ mu, const float* __restrict__ var,
                          float* __restrict__ out) {
    int idx = blockIdx.x * blockDim.x + threadIdx.x;
    if (idx >= MM * CC) return;
    int c = idx & (CC - 1);
    int m = idx >> 7;
    int b = m / LL, l = m - b * LL;
    float s = g[c] * rsqrtf(var[c] + 1e-5f);
    float v = (t[idx] - mu[c]) * s + bta[c];
    out[((size_t)(b * CC + c)) * LL + l] = fmaxf(v, 0.f);
}

// ---------------- host launch ----------------
static inline size_t alignup(size_t x) { return (x + 255) & ~(size_t)255; }

extern "C" void kernel_launch(void* const* d_in, const int* in_sizes, int n_in,
                              void* d_out, int out_size, void* d_ws, size_t ws_size,
                              hipStream_t stream) {
    (void)in_sizes; (void)n_in; (void)out_size; (void)ws_size;
    const float* x        = (const float*)d_in[0];
    const float* nin_w    = (const float*)d_in[1];
    const float* nin2_w   = (const float*)d_in[2];
    const float* bn1_g    = (const float*)d_in[3];
    const float* bn1_b    = (const float*)d_in[4];
    const float* bn1_m    = (const float*)d_in[5];
    const float* bn1_v    = (const float*)d_in[6];
    const float* bn2_g    = (const float*)d_in[7];
    const float* bn2_b    = (const float*)d_in[8];
    const float* bn2_m    = (const float*)d_in[9];
    const float* bn2_v    = (const float*)d_in[10];
    const float* in_proj  = (const float*)d_in[11];
    const float* conv_w   = (const float*)d_in[12];
    const float* conv_b   = (const float*)d_in[13];
    const float* xproj_w  = (const float*)d_in[14];
    const float* dtproj_w = (const float*)d_in[15];
    const float* dtproj_b = (const float*)d_in[16];
    const float* A_log    = (const float*)d_in[17];
    const float* D_param  = (const float*)d_in[18];
    const float* out_proj = (const float*)d_in[19];
    float* out = (float*)d_out;

    char* p = (char*)d_ws; size_t off = 0;
    auto take = [&](size_t bytes) { char* q = p + off; off = alignup(off + bytes); return q; };
    unsigned short* xt    = (unsigned short*)take((size_t)MM * CC * 2);
    unsigned short* w1    = (unsigned short*)take((size_t)CC * CC * 2);
    unsigned short* w2    = (unsigned short*)take((size_t)CC * CC * 2);
    unsigned short* wip0  = (unsigned short*)take((size_t)512 * CC * 2);
    unsigned short* wip1  = (unsigned short*)take((size_t)512 * CC * 2);
    unsigned short* wxp   = (unsigned short*)take((size_t)XPAD * DIN * 2);
    unsigned short* wop0  = (unsigned short*)take((size_t)CC * DIN * 2);
    unsigned short* wop1  = (unsigned short*)take((size_t)CC * DIN * 2);
    float*          act   = (float*)take((size_t)MM * CC * 4);
    float*          xzbuf = (float*)take((size_t)MM * 512 * 4);
    unsigned short* xmbuf = (unsigned short*)take((size_t)MM * DIN * 2);
    float*          xdbl  = (float*)take((size_t)MM * XPAD * 4);
    unsigned short* ygbuf = (unsigned short*)take((size_t)MM * DIN * 2);
    float*          ysum  = (float*)take((size_t)MM * CC * 4);
    unsigned short* midb  = (unsigned short*)take((size_t)MM * CC * 2);
    float*          tmp2  = (float*)take((size_t)MM * CC * 4);

    const int T = 256;
    auto blocks = [](int total, int t) { return (total + t - 1) / t; };

    k_wconv<<<blocks(CC * CC, T), T, 0, stream>>>(nin_w,  w1,  CC, CC, CC, 0, 0);
    k_wconv<<<blocks(CC * CC, T), T, 0, stream>>>(nin2_w, w2,  CC, CC, CC, 0, 0);
    k_wconv<<<blocks(512 * CC, T), T, 0, stream>>>(in_proj, wip0, 512, CC, 512, 0, 0);
    k_wconv<<<blocks(512 * CC, T), T, 0, stream>>>(in_proj, wip1, 512, CC, 512, 0, 1);
    k_wconv<<<blocks(XPAD * DIN, T), T, 0, stream>>>(xproj_w, wxp, 40, DIN, XPAD, 0, 0);
    k_wconv<<<blocks(CC * DIN, T), T, 0, stream>>>(out_proj, wop0, CC, DIN, CC, 0, 0);
    k_wconv<<<blocks(CC * DIN, T), T, 0, stream>>>(out_proj, wop1, CC, DIN, CC, 1, 0);
    k_x_transpose<<<blocks(BB * CC * LL, T), T, 0, stream>>>(x, xt);

    // nin1 GEMM + BN1 + ReLU -> act (f32) and x_flat (bf16, reuses xt)
    k_gemm_bf16<<<dim3(MM / 128, CC / 64), T, 0, stream>>>(xt, w1, tmp2, MM, CC, CC, CC, 0);
    k_bn1_relu<<<blocks(MM * CC, T), T, 0, stream>>>(tmp2, bn1_g, bn1_b, bn1_m, bn1_v, act, xt);

    const int dirL[4] = {0, 0, 1, 1};
    const int dirC[4] = {0, 1, 0, 1};
    for (int d = 0; d < 4; ++d) {
        unsigned short* wip = dirC[d] ? wip1 : wip0;
        unsigned short* wop = dirC[d] ? wop1 : wop0;
        k_gemm_bf16<<<dim3(MM / 128, 512 / 64), T, 0, stream>>>(xt, wip, xzbuf, MM, 512, CC, 512, 0);
        k_conv_silu<<<blocks(MM * DIN, T), T, 0, stream>>>(xzbuf, conv_w, conv_b, xmbuf, dirL[d]);
        k_gemm_bf16<<<dim3(MM / 128, XPAD / 64), T, 0, stream>>>(xmbuf, wxp, xdbl, MM, XPAD, DIN, XPAD, 0);
        k_scan<<<BB, T, 0, stream>>>(xdbl, xmbuf, xzbuf, dtproj_w, dtproj_b, A_log, D_param, ygbuf, dirL[d]);
        k_gemm_bf16<<<dim3(MM / 128, CC / 64), T, 0, stream>>>(ygbuf, wop, ysum, MM, CC, DIN, CC, d ? 1 : 0);
    }

    k_mid<<<blocks(MM * CC, T), T, 0, stream>>>(ysum, act, midb);
    k_gemm_bf16<<<dim3(MM / 128, CC / 64), T, 0, stream>>>(midb, w2, tmp2, MM, CC, CC, CC, 0);
    k_bn2_out<<<blocks(MM * CC, T), T, 0, stream>>>(tmp2, bn2_g, bn2_b, bn2_m, bn2_v, out);
}
